// Attention_3539053052511
// MI455X (gfx1250) — matrix operand order repaired and checked by reading, not compiled, not run on hardware
//
#include <hip/hip_runtime.h>
#include <hip/hip_bf16.h>
#include <stdint.h>

// ---------------------------------------------------------------------------
// MI455X / gfx1250 attention block. All matmuls on V_WMMA_F32_16X16X32_BF16,
// shared tiles staged through LDS with GLOBAL_LOAD_ASYNC_TO_LDS_B128
// (ASYNCcnt double buffering). Fragments grouped: bulk ds_load, then WMMAs
// issued back-to-back.
// ---------------------------------------------------------------------------

typedef __attribute__((ext_vector_type(16))) __bf16 v16bf;
typedef __attribute__((ext_vector_type(8)))  float  v8f;

union Frag16 { v16bf v; uint32_t u[8]; uint4 q[2]; };

__device__ __forceinline__ uint16_t f2bf(float f) {
  uint32_t x = __builtin_bit_cast(uint32_t, f);
  uint32_t r = x + 0x7FFFu + ((x >> 16) & 1u);   // round-to-nearest-even
  return (uint16_t)(r >> 16);
}
__device__ __forceinline__ uint32_t pack_bf(float a, float b) {
  return (uint32_t)f2bf(a) | ((uint32_t)f2bf(b) << 16);
}

// Raw 32-bit LDS offset of a __shared__ object (addrspace(3) ptrtoint).
typedef __attribute__((address_space(3))) uint16_t lds_u16;
__device__ __forceinline__ uint32_t lds_off(const uint16_t* p) {
  return (uint32_t)(uintptr_t)(lds_u16*)p;
}

// Async DMA: 16 bytes global -> LDS, tracked by ASYNCcnt.
__device__ __forceinline__ void async_ld_b128(uint32_t lds_addr,
                                              const uint16_t* gptr) {
  asm volatile("global_load_async_to_lds_b128 %0, %1, off"
               :: "v"(lds_addr), "v"(gptr) : "memory");
}
__device__ __forceinline__ void wait_async_le2() {
  asm volatile("s_wait_asynccnt 0x2" ::: "memory");
}
__device__ __forceinline__ void wait_async_le4() {
  asm volatile("s_wait_asynccnt 0x4" ::: "memory");
}
__device__ __forceinline__ void wait_ds0() {
  asm volatile("s_wait_dscnt 0x0" ::: "memory");
}

#define WMMA_BF16(A, B, C)                                                    \
  __builtin_amdgcn_wmma_f32_16x16x32_bf16(false, (A).v, false, (B).v,         \
                                          (short)0, (C), false, false)

// ---------------------------------------------------------------------------
// Kernel 1: f32 -> bf16 conversion (grid-stride)
// ---------------------------------------------------------------------------
__global__ void cvt_f32_bf16(const float* __restrict__ src,
                             uint16_t* __restrict__ dst, int n) {
  int i = blockIdx.x * blockDim.x + threadIdx.x;
  int stride = gridDim.x * blockDim.x;
  for (; i < n; i += stride) dst[i] = f2bf(src[i]);
}

// ---------------------------------------------------------------------------
// Kernel 2/5: C[M,Nc] = A[M,K] @ Bw[Nc,K]^T (+bias), bf16 in / f32 out.
// 256 threads = 8 waves. Block tile 256x64, wave tile 32x64 (8 accums).
// B tile (64 cols x 64 K = 8KB) async-staged in LDS, double buffered:
// 16 WMMAs per barrier pair.
// ---------------------------------------------------------------------------
__global__ void gemm_bf16_wmma(const uint16_t* __restrict__ A,
                               const uint16_t* __restrict__ Bw,
                               float* __restrict__ C,
                               const float* __restrict__ bias,
                               int M, int Nc, int K) {
  __shared__ __align__(16) uint16_t Bt[2][64][64];   // 16 KB

  const int tid  = threadIdx.x;
  const int lane = tid & 31;
  const int wave = tid >> 5;
  const int half = lane >> 4;
  const int l16  = lane & 15;
  const int row0 = blockIdx.x * 256 + wave * 32;
  const int col0 = blockIdx.y * 64;

  // staging: thread t moves 32B of one column: col = t/4, offs = (2t&7)*8
  const int stc = tid >> 2;
  const int sto = ((tid * 2) & 7) * 8;
  const uint16_t* Bsrc = Bw + (size_t)(col0 + stc) * K + sto;

  const uint16_t* Arow0 = A + (size_t)(row0 + l16) * K;
  const uint16_t* Arow1 = Arow0 + (size_t)16 * K;

  v8f acc[8] = {};

  // prologue: K-tile 0 into buf 0
  async_ld_b128(lds_off(&Bt[0][stc][sto]), Bsrc);
  async_ld_b128(lds_off(&Bt[0][stc][sto + 8]), Bsrc + 8);

  for (int k0 = 0; k0 < K; k0 += 64) {
    const int buf = (k0 >> 6) & 1;
    __syncthreads();   // readers of buf^1 (prev iter) are done
    const int kn = (k0 + 64 < K) ? (k0 + 64) : 0;   // wrap: harmless re-read
    async_ld_b128(lds_off(&Bt[buf ^ 1][stc][sto]), Bsrc + kn);
    async_ld_b128(lds_off(&Bt[buf ^ 1][stc][sto + 8]), Bsrc + kn + 8);
    wait_async_le2();  // current tile landed; next stays in flight
    __syncthreads();   // publish LDS writes to all waves

    __builtin_prefetch(Arow0 + k0 + 256, 0, 3);
#pragma unroll
    for (int ks = 0; ks < 64; ks += 32) {
      // ---- bulk fragment loads ----
      Frag16 a0, a1;
      a0.q[0] = *(const uint4*)(Arow0 + k0 + ks + half * 8);
      a0.q[1] = *(const uint4*)(Arow0 + k0 + ks + 16 + half * 8);
      a1.q[0] = *(const uint4*)(Arow1 + k0 + ks + half * 8);
      a1.q[1] = *(const uint4*)(Arow1 + k0 + ks + 16 + half * 8);
      Frag16 bfr[4];
#pragma unroll
      for (int g = 0; g < 4; ++g) {
        const uint16_t* bp = &Bt[buf][g * 16 + l16][ks + half * 8];
        bfr[g].q[0] = *(const uint4*)(bp);
        bfr[g].q[1] = *(const uint4*)(bp + 16);
      }
      // ---- WMMAs back-to-back ----
#pragma unroll
      for (int g = 0; g < 4; ++g) acc[g]     = WMMA_BF16(a0, bfr[g], acc[g]);
#pragma unroll
      for (int g = 0; g < 4; ++g) acc[4 + g] = WMMA_BF16(a1, bfr[g], acc[4 + g]);
    }
  }

#pragma unroll
  for (int g = 0; g < 4; ++g) {
    const int col = col0 + g * 16 + l16;
    const float bv = bias ? bias[col] : 0.f;
#pragma unroll
    for (int v = 0; v < 8; ++v) {
      const int rr = row0 + v + half * 8;
      C[(size_t)rr * Nc + col]        = acc[g][v] + bv;
      C[(size_t)(rr + 16) * Nc + col] = acc[4 + g][v] + bv;
    }
  }
}

// ---------------------------------------------------------------------------
// Kernel 3: RMSNorm(q,k) + rotary + repack.
// qkvf: [B*N, 3072] f32.  Qb/Kb: [BH, N, 64] bf16.  Vt: [BH, 64, N] bf16.
// ---------------------------------------------------------------------------
__global__ void norm_rope_pack(const float* __restrict__ qkvf,
                               const float* __restrict__ pos,
                               const float* __restrict__ qw,
                               const float* __restrict__ kw,
                               uint16_t* __restrict__ Qb,
                               uint16_t* __restrict__ Kb,
                               uint16_t* __restrict__ Vt) {
  const int lane = threadIdx.x & 31;
  const int wave = threadIdx.x >> 5;
  const int r  = blockIdx.x * (blockDim.x >> 5) + wave;   // 0..65535
  const int n  = r & 2047;
  const int bh = r >> 11;
  const int b  = bh >> 4;
  const int h  = bh & 15;
  const int d0 = lane * 2, d1 = d0 + 1;
  const size_t base = ((size_t)(b * 2048 + n)) * 3072 + (size_t)h * 64;

  float sn, cn;
  __sincosf(pos[n * 32 + lane], &sn, &cn);

  {  // Q
    float x0 = qkvf[base + d0], x1 = qkvf[base + d1];
    float ss = x0 * x0 + x1 * x1;
#pragma unroll
    for (int m = 16; m >= 1; m >>= 1) ss += __shfl_xor(ss, m, 32);
    float rms = rsqrtf(ss * (1.f / 64.f) + 1e-6f);
    float xr = x0 * rms * qw[d0], xi = x1 * rms * qw[d1];
    *(uint32_t*)(Qb + ((size_t)bh * 2048 + n) * 64 + d0) =
        pack_bf(xr * cn - xi * sn, xr * sn + xi * cn);
  }
  {  // K
    float x0 = qkvf[base + 1024 + d0], x1 = qkvf[base + 1024 + d1];
    float ss = x0 * x0 + x1 * x1;
#pragma unroll
    for (int m = 16; m >= 1; m >>= 1) ss += __shfl_xor(ss, m, 32);
    float rms = rsqrtf(ss * (1.f / 64.f) + 1e-6f);
    float xr = x0 * rms * kw[d0], xi = x1 * rms * kw[d1];
    *(uint32_t*)(Kb + ((size_t)bh * 2048 + n) * 64 + d0) =
        pack_bf(xr * cn - xi * sn, xr * sn + xi * cn);
  }
  {  // V transposed [bh, d, n]
    float v0 = qkvf[base + 2048 + d0], v1 = qkvf[base + 2048 + d1];
    Vt[((size_t)bh * 64 + d0) * 2048 + n] = f2bf(v0);
    Vt[((size_t)bh * 64 + d1) * 2048 + n] = f2bf(v1);
  }
}

// ---------------------------------------------------------------------------
// Kernel 4: flash-style attention. Block = 4 waves, each wave 16 query rows,
// KV tiles of 32. K / V^T tiles async-staged in LDS (double buffered).
// P converts D-layout -> A-layout through per-wave LDS.
// ---------------------------------------------------------------------------
__global__ void attn_wmma(const uint16_t* __restrict__ Qb,
                          const uint16_t* __restrict__ Kb,
                          const uint16_t* __restrict__ Vt,
                          uint16_t* __restrict__ Obuf) {
  __shared__ __align__(16) uint16_t Kt[2][32][64];   // [key][d]      8 KB
  __shared__ __align__(16) uint16_t Vl[2][64][32];   // [d][kv]       8 KB
  __shared__ __align__(16) uint16_t Pl[4][16][32];   // per-wave P    4 KB

  const int tid  = threadIdx.x;        // 128 threads
  const int lane = tid & 31;
  const int wave = tid >> 5;
  const int half = lane >> 4;
  const int l16  = lane & 15;
  const int bh = blockIdx.y;
  const int b  = bh >> 4;
  const int h  = bh & 15;
  const int q0 = blockIdx.x * 64 + wave * 16;

  // --- async staging geometry: 256 x 16B segs per tile, 2 per thread -----
  const int ks0 = tid * 2, ks1 = tid * 2 + 1;
  const int kKey0 = ks0 >> 3, kOff0 = (ks0 & 7) * 8;
  const int kKey1 = ks1 >> 3, kOff1 = (ks1 & 7) * 8;
  const int vD0 = ks0 >> 2, vOff0 = (ks0 & 3) * 8;
  const int vD1 = ks1 >> 2, vOff1 = (ks1 & 3) * 8;
  const uint16_t* Kbase = Kb + (size_t)bh * 2048 * 64;
  const uint16_t* Vbase = Vt + (size_t)bh * 64 * 2048;

#define STAGE_KV(bufi, kv)                                                   \
  do {                                                                       \
    async_ld_b128(lds_off(&Kt[bufi][kKey0][kOff0]),                          \
                  Kbase + (size_t)((kv) + kKey0) * 64 + kOff0);              \
    async_ld_b128(lds_off(&Kt[bufi][kKey1][kOff1]),                          \
                  Kbase + (size_t)((kv) + kKey1) * 64 + kOff1);              \
    async_ld_b128(lds_off(&Vl[bufi][vD0][vOff0]),                            \
                  Vbase + (size_t)vD0 * 2048 + (kv) + vOff0);                \
    async_ld_b128(lds_off(&Vl[bufi][vD1][vOff1]),                            \
                  Vbase + (size_t)vD1 * 2048 + (kv) + vOff1);                \
  } while (0)

  // --- Q fragments (16 x 64 = two 16x32 A-frags), loaded once -----------
  Frag16 qf[2];
  {
    const uint16_t* Qrow = Qb + ((size_t)bh * 2048 + q0 + l16) * 64;
#pragma unroll
    for (int c = 0; c < 2; ++c) {
      qf[c].q[0] = *(const uint4*)(Qrow + c * 32 + half * 8);
      qf[c].q[1] = *(const uint4*)(Qrow + c * 32 + 16 + half * 8);
    }
  }

  float mi[8], li[8];
#pragma unroll
  for (int v = 0; v < 8; ++v) { mi[v] = -1e30f; li[v] = 0.f; }
  v8f acc[4] = {};
  const float scale = 0.125f;   // 64^-0.5

  STAGE_KV(0, 0);   // prologue

  for (int kv0 = 0; kv0 < 2048; kv0 += 32) {
    const int buf = (kv0 >> 5) & 1;
    __syncthreads();                       // buf^1 readers (prev iter) done
    STAGE_KV(buf ^ 1, (kv0 + 32) & 2047);  // wrap: harmless re-read
    wait_async_le4();                      // current tile landed
    __syncthreads();                       // publish to all waves

    // ---- S = scale * Q @ K^T : bulk loads, then 4 WMMAs ----
    Frag16 kb[2][2];
#pragma unroll
    for (int t = 0; t < 2; ++t)
#pragma unroll
      for (int c = 0; c < 2; ++c) {
        const uint16_t* kp = &Kt[buf][t * 16 + l16][c * 32 + half * 8];
        kb[t][c].q[0] = *(const uint4*)(kp);
        kb[t][c].q[1] = *(const uint4*)(kp + 16);
      }
    v8f sacc[2] = {};
    sacc[0] = WMMA_BF16(qf[0], kb[0][0], sacc[0]);
    sacc[1] = WMMA_BF16(qf[0], kb[1][0], sacc[1]);
    sacc[0] = WMMA_BF16(qf[1], kb[0][1], sacc[0]);
    sacc[1] = WMMA_BF16(qf[1], kb[1][1], sacc[1]);

    // ---- online softmax (rows live across 16-lane halves) ----
    float alpha[8];
#pragma unroll
    for (int v = 0; v < 8; ++v) {
      float s0 = sacc[0][v] * scale;
      float s1 = sacc[1][v] * scale;
      float mx = fmaxf(s0, s1);
#pragma unroll
      for (int msk = 1; msk < 16; msk <<= 1)
        mx = fmaxf(mx, __shfl_xor(mx, msk, 32));
      float mnew = fmaxf(mi[v], mx);
      alpha[v] = __expf(mi[v] - mnew);
      float p0 = __expf(s0 - mnew);
      float p1 = __expf(s1 - mnew);
      sacc[0][v] = p0; sacc[1][v] = p1;
      float rs = p0 + p1;
#pragma unroll
      for (int msk = 1; msk < 16; msk <<= 1) rs += __shfl_xor(rs, msk, 32);
      li[v] = li[v] * alpha[v] + rs;
      mi[v] = mnew;
    }
    // ---- P: D-layout -> LDS (bf16, row-major 16x32) -> A-layout ----
#pragma unroll
    for (int v = 0; v < 8; ++v) {
      const int row = v + half * 8;
      Pl[wave][row][l16]      = f2bf(sacc[0][v]);
      Pl[wave][row][16 + l16] = f2bf(sacc[1][v]);
    }
    wait_ds0();
    Frag16 pf;
    pf.q[0] = *(const uint4*)&Pl[wave][l16][half * 8];
    pf.q[1] = *(const uint4*)&Pl[wave][l16][16 + half * 8];

    // ---- bulk V loads, rescale, then 4 WMMAs ----
    Frag16 vf[4];
#pragma unroll
    for (int g = 0; g < 4; ++g) {
      const uint16_t* vp = &Vl[buf][g * 16 + l16][half * 8];
      vf[g].q[0] = *(const uint4*)(vp);
      vf[g].q[1] = *(const uint4*)(vp + 16);
    }
#pragma unroll
    for (int g = 0; g < 4; ++g)
#pragma unroll
      for (int v = 0; v < 8; ++v) acc[g][v] *= alpha[v];
#pragma unroll
    for (int g = 0; g < 4; ++g) acc[g] = WMMA_BF16(pf, vf[g], acc[g]);
  }
#undef STAGE_KV

  // ---- epilogue: normalize, scatter bf16 into [B*N, 1024] ----
#pragma unroll
  for (int v = 0; v < 8; ++v) li[v] = 1.f / li[v];
#pragma unroll
  for (int g = 0; g < 4; ++g) {
    const int col = h * 64 + g * 16 + l16;
#pragma unroll
    for (int v = 0; v < 8; ++v) {
      const int row = q0 + v + half * 8;
      Obuf[((size_t)b * 2048 + row) * 1024 + col] = f2bf(acc[g][v] * li[v]);
    }
  }
}

// ---------------------------------------------------------------------------
// Host-side launch
// ---------------------------------------------------------------------------
extern "C" void kernel_launch(void* const* d_in, const int* in_sizes, int n_in,
                              void* d_out, int out_size, void* d_ws,
                              size_t ws_size, hipStream_t stream) {
  (void)in_sizes; (void)n_in; (void)out_size; (void)ws_size;
  const float* x      = (const float*)d_in[0];   // [2,2048,1024]
  const float* pos    = (const float*)d_in[1];   // [2048,32]
  const float* qkv_w  = (const float*)d_in[2];   // [3072,1024]
  const float* q_norm = (const float*)d_in[3];   // [64]
  const float* k_norm = (const float*)d_in[4];   // [64]
  const float* proj_w = (const float*)d_in[5];   // [1024,1024]
  const float* proj_b = (const float*)d_in[6];   // [1024]
  float* out = (float*)d_out;                    // [2,2048,1024] f32

  char* ws = (char*)d_ws;
  uint16_t* xb    = (uint16_t*)(ws);                               //  8 MB
  uint16_t* wqkv  = (uint16_t*)(ws + ( 8ull << 20));               //  6 MB
  uint16_t* wproj = (uint16_t*)(ws + (14ull << 20));               //  2 MB
  float*    qkvf  = (float*)   (ws + (16ull << 20));               // 48 MB
  uint16_t* Qb    = (uint16_t*)(ws + (64ull << 20));               //  8 MB
  uint16_t* Kb    = (uint16_t*)(ws + (72ull << 20));               //  8 MB
  uint16_t* Vt    = (uint16_t*)(ws + (80ull << 20));               //  8 MB
  uint16_t* Obuf  = (uint16_t*)(ws + (88ull << 20));               //  8 MB

  cvt_f32_bf16<<<1024, 256, 0, stream>>>(x,      xb,    4096 * 1024);
  cvt_f32_bf16<<<1024, 256, 0, stream>>>(qkv_w,  wqkv,  3072 * 1024);
  cvt_f32_bf16<<< 512, 256, 0, stream>>>(proj_w, wproj, 1024 * 1024);

  gemm_bf16_wmma<<<dim3(4096 / 256, 3072 / 64), 256, 0, stream>>>(
      xb, wqkv, qkvf, nullptr, 4096, 3072, 1024);

  norm_rope_pack<<<65536 / 8, 256, 0, stream>>>(qkvf, pos, q_norm, k_norm,
                                                Qb, Kb, Vt);

  attn_wmma<<<dim3(2048 / 64, 32), 128, 0, stream>>>(Qb, Kb, Vt, Obuf);

  gemm_bf16_wmma<<<dim3(4096 / 256, 1024 / 64), 256, 0, stream>>>(
      Obuf, wproj, out, proj_b, 4096, 1024, 1024);
}
